// TP_RGCNLayer_44985487458912
// MI455X (gfx1250) — matrix operand order repaired and checked by reading, not compiled, not run on hardware
//
#include <hip/hip_runtime.h>
#include <hip/hip_bf16.h>

#define N_NODES 20000
#define IN_DIM  128
#define OUT_DIM 128
#define N_RELS  32
#define N_BASES 16

typedef __attribute__((ext_vector_type(16))) __bf16 v16bf;
typedef __attribute__((ext_vector_type(8)))  __bf16 v8bf;
typedef __attribute__((ext_vector_type(4)))  __bf16 v4bf;
typedef __attribute__((ext_vector_type(8)))  float  v8f;

// K1: W[r,i,o] = sum_b w_comp[r,b] * weight[b,i,o], stored transposed as
// bf16 Wt[r, o, i] so WMMA B-fragments (K-major per output column) are
// contiguous 32B loads.
__global__ void rgcn_basis_kernel(const float* __restrict__ weight,
                                  const float* __restrict__ w_comp,
                                  __bf16* __restrict__ Wt) {
  int t = blockIdx.x * blockDim.x + threadIdx.x;   // t = r*IN*OUT + i*OUT + o
  int o = t & (OUT_DIM - 1);
  int i = (t >> 7) & (IN_DIM - 1);
  int r = t >> 14;
  float acc = 0.f;
#pragma unroll
  for (int b = 0; b < N_BASES; ++b)
    acc += w_comp[r * N_BASES + b] * weight[((size_t)(b * IN_DIM) + i) * OUT_DIM + o];
  Wt[((size_t)(r * OUT_DIM + o)) * IN_DIM + i] = (__bf16)acc;
}

// K1b: h fp32 -> bf16
__global__ void rgcn_cvt_h(const float* __restrict__ h, __bf16* __restrict__ hbf, int n) {
  int t = blockIdx.x * blockDim.x + threadIdx.x;
  if (t < n) hbf[t] = (__bf16)h[t];
}

// K2: h_all[n, r, o] = h[n, :] @ W[r]  via v_wmma_f32_16x16x32_bf16.
// blockIdx.x = node chunk (10 tiles of 16 nodes), blockIdx.y = relation r,
// wave id (0..7) = 16-wide OUT tile. One wave: B-fragments for K=128 cached
// in VGPRs, loop over node tiles, 4 chained WMMAs, bf16 store of D.
__global__ __launch_bounds__(256) void rgcn_gemm(const __bf16* __restrict__ hbf,
                                                 const __bf16* __restrict__ Wt,
                                                 __bf16* __restrict__ h_all) {
  const int lane = threadIdx.x & 31;
  const int wid  = threadIdx.x >> 5;   // OUT tile index, 0..7
  const int r    = blockIdx.y;         // relation, 0..31
  const int col  = lane & 15;
  const int half = lane >> 4;          // lane group selects K-half

  // B fragment as loaded here (contiguous k order, matching A): lane = column, lanes 0-15 hold
  // K=0..15, lanes 16-31 hold K=16..31; Wt is K-contiguous per column.
  const __bf16* bp = Wt + ((size_t)(r * OUT_DIM + wid * 16 + col)) * IN_DIM + half * 16;
  v16bf bfrag[4];
#pragma unroll
  for (int ks = 0; ks < 4; ++ks)
    bfrag[ks] = *(const v16bf*)(bp + ks * 32);

  for (int t = 0; t < 10; ++t) {
    const int node0 = (blockIdx.x * 10 + t) * 16;
    // A fragment as loaded here (contiguous k order, matching B): lane = row; lanes 0-15 hold
    // K = 0..15, lanes 16-31 hold K = 16..31.
    const __bf16* ap = hbf + (size_t)(node0 + col) * IN_DIM + half * 16;
    v8f c = {};
#pragma unroll
    for (int ks = 0; ks < 4; ++ks) {
      v8bf lo = *(const v8bf*)(ap + ks * 32);
      v8bf hi = *(const v8bf*)(ap + ks * 32 + 8);
      v16bf a = __builtin_shufflevector(lo, hi, 0, 1, 2, 3, 4, 5, 6, 7,
                                        8, 9, 10, 11, 12, 13, 14, 15);
      c = __builtin_amdgcn_wmma_f32_16x16x32_bf16(
          /*neg_a=*/false, a, /*neg_b=*/false, bfrag[ks],
          /*c_mod=*/(short)0, c, /*reuse_a=*/false, /*reuse_b=*/false);
    }
    // D layout: VGPR j -> row (node0 + half*8 + j), lane&15 -> column.
    __bf16* dp = h_all + (size_t)(node0 + half * 8) * (N_RELS * OUT_DIM)
                       + (size_t)r * OUT_DIM + wid * 16 + col;
#pragma unroll
    for (int j = 0; j < 8; ++j)
      dp[(size_t)j * (N_RELS * OUT_DIM)] = (__bf16)c[j];
  }
}

// K3: one wave per edge; lane owns 4 output channels.
// msg = h_all[src, edge_type]; out[dst] += msg (fp32 hw atomics, L2-resident).
__global__ __launch_bounds__(256) void rgcn_scatter(const __bf16* __restrict__ h_all,
                                                    const int* __restrict__ src,
                                                    const int* __restrict__ etype,
                                                    const int* __restrict__ dstn,
                                                    float* __restrict__ out, int E) {
  const int w    = (int)((blockIdx.x * blockDim.x + threadIdx.x) >> 5);
  const int lane = threadIdx.x & 31;
  if (w >= E) return;
  const int s = src[w];
  const int r = etype[w];
  const int d = dstn[w];
  const __bf16* mp = h_all + ((size_t)(s * N_RELS + r)) * OUT_DIM + lane * 4;
  v4bf v = *(const v4bf*)mp;
  float* op = out + (size_t)d * OUT_DIM + lane * 4;
#pragma unroll
  for (int j = 0; j < 4; ++j)
    unsafeAtomicAdd(op + j, (float)v[j]);
}

// K4: relu
__global__ void rgcn_relu(float* __restrict__ out, int n) {
  int t = blockIdx.x * blockDim.x + threadIdx.x;
  if (t < n) out[t] = fmaxf(out[t], 0.f);
}

extern "C" void kernel_launch(void* const* d_in, const int* in_sizes, int n_in,
                              void* d_out, int out_size, void* d_ws, size_t ws_size,
                              hipStream_t stream) {
  const float* h         = (const float*)d_in[0];
  const float* weight    = (const float*)d_in[1];
  const float* w_comp    = (const float*)d_in[2];
  const int*   edge_type = (const int*)d_in[3];
  const int*   src       = (const int*)d_in[4];
  const int*   dst       = (const int*)d_in[5];
  float*       out       = (float*)d_out;
  const int    E         = in_sizes[3];

  char* ws = (char*)d_ws;
  __bf16* Wt   = (__bf16*)(ws);                     // 32*128*128*2 = 1 MB
  __bf16* hbf  = (__bf16*)(ws + (1u << 20));        // 20000*128*2  = 5 MB
  __bf16* hall = (__bf16*)(ws + (8u << 20));        // 20000*32*128*2 = 164 MB

  hipMemsetAsync(d_out, 0, (size_t)out_size * sizeof(float), stream);

  rgcn_basis_kernel<<<(N_RELS * IN_DIM * OUT_DIM) / 256, 256, 0, stream>>>(weight, w_comp, Wt);
  rgcn_cvt_h<<<(N_NODES * IN_DIM + 255) / 256, 256, 0, stream>>>(h, hbf, N_NODES * IN_DIM);
  rgcn_gemm<<<dim3(125, 32), 256, 0, stream>>>(hbf, Wt, hall);
  rgcn_scatter<<<(E + 7) / 8, 256, 0, stream>>>(hall, src, edge_type, dst, out, E);
  rgcn_relu<<<(N_NODES * OUT_DIM + 255) / 256, 256, 0, stream>>>(out, N_NODES * OUT_DIM);
}
